// Head_60335700574591
// MI455X (gfx1250) — compile-verified, matrix operand order checked by reading, not run on hardware
//
#include <hip/hip_runtime.h>

#define B_BATCH 4
#define T_SEQ   4096
#define E_DIM   1024
#define D_DIM   64

typedef __attribute__((ext_vector_type(16))) __bf16        bf16x16;
typedef __attribute__((ext_vector_type(8)))  float         f32x8;
typedef __attribute__((ext_vector_type(8)))  unsigned int  u32x8;

union BF16Vec { u32x8 u; bf16x16 b; };

// Native f32 -> bf16 (RNE) via hardware convert path.
__device__ __forceinline__ unsigned short f2bf(float f) {
  return __builtin_bit_cast(unsigned short, (__bf16)f);
}

__device__ __forceinline__ f32x8 wmma_bf16(bf16x16 a, bf16x16 b, f32x8 c) {
  // D = A(16x32 bf16) x B(32x16 bf16) + C(16x16 f32)
  return __builtin_amdgcn_wmma_f32_16x16x32_bf16(false, a, false, b, (short)0, c,
                                                 false, false);
}

// Assemble a 16-element bf16 operand from two contiguous 16-byte chunks.
__device__ __forceinline__ bf16x16 ld16(const unsigned short* p0,
                                        const unsigned short* p1) {
  BF16Vec v;
  uint4 a = *(const uint4*)p0;
  uint4 b = *(const uint4*)p1;
  v.u[0] = a.x; v.u[1] = a.y; v.u[2] = a.z; v.u[3] = a.w;
  v.u[4] = b.x; v.u[5] = b.y; v.u[6] = b.z; v.u[7] = b.w;
  return v.b;
}

// Same, but from fp32 memory with native conversion to bf16.
__device__ __forceinline__ bf16x16 cvt16(const float* p0, const float* p1) {
  float4 a0 = ((const float4*)p0)[0];
  float4 a1 = ((const float4*)p0)[1];
  float4 b0 = ((const float4*)p1)[0];
  float4 b1 = ((const float4*)p1)[1];
  bf16x16 v;
  v[0]  = (__bf16)a0.x; v[1]  = (__bf16)a0.y;
  v[2]  = (__bf16)a0.z; v[3]  = (__bf16)a0.w;
  v[4]  = (__bf16)a1.x; v[5]  = (__bf16)a1.y;
  v[6]  = (__bf16)a1.z; v[7]  = (__bf16)a1.w;
  v[8]  = (__bf16)b0.x; v[9]  = (__bf16)b0.y;
  v[10] = (__bf16)b0.z; v[11] = (__bf16)b0.w;
  v[12] = (__bf16)b1.x; v[13] = (__bf16)b1.y;
  v[14] = (__bf16)b1.z; v[15] = (__bf16)b1.w;
  return v;
}

// CDNA5 async DMA: copy 16B global -> LDS, tracked by ASYNCcnt (no VGPR data).
__device__ __forceinline__ void async_ld_b128(unsigned lds_off, const void* gptr) {
  asm volatile("global_load_async_to_lds_b128 %0, %1, off"
               :: "v"(lds_off), "v"((unsigned long long)(size_t)gptr)
               : "memory");
}

// ---------------------------------------------------------------------------
// Kernel 1: QKV projection.  y = x * W + bias, x:[16384,1024] f32, W:[1024,64].
// blockIdx.y selects q/k/v.  q,k stored row-major bf16; v stored transposed
// (vT[B][64][T]) so the attention kernel's V^T A operands are contiguous.
// ---------------------------------------------------------------------------
__global__ __launch_bounds__(256) void qkv_proj_kernel(
    const float* __restrict__ x,
    const float* __restrict__ Wq, const float* __restrict__ bq,
    const float* __restrict__ Wk, const float* __restrict__ bk,
    const float* __restrict__ Wv, const float* __restrict__ bv,
    unsigned short* __restrict__ qo, unsigned short* __restrict__ ko,
    unsigned short* __restrict__ vto)
{
  const int which = blockIdx.y;
  const float* W    = (which == 0) ? Wq : (which == 1) ? Wk : Wv;
  const float* bias = (which == 0) ? bq : (which == 1) ? bk : bv;

  const int tid  = threadIdx.x;
  const int wave = tid >> 5;
  const int lane = tid & 31;
  const int hf   = lane >> 4;       // lane half (WMMA layout)
  const int lq   = lane & 15;
  const int kb   = hf ? 8 : 0;      // K-base for A/B operand halves

  const int mbase   = blockIdx.x * 128;
  const int rowbase = mbase + wave * 16;
  const int arow    = rowbase + lq;

  __shared__ __align__(16) unsigned short Wl[64 * 32];  // [col][kk] bf16

  f32x8 zero8 = {0.f, 0.f, 0.f, 0.f, 0.f, 0.f, 0.f, 0.f};
  f32x8 acc[4] = {zero8, zero8, zero8, zero8};

  const int idx0 = tid * 8;         // staging: 8 consecutive W elements/thread
  const int scol = idx0 & 63;
  const int skk  = idx0 >> 6;

  for (int k0 = 0; k0 < E_DIM; k0 += 32) {
    __syncthreads();
    {
      const float4* wp = (const float4*)&W[(k0 + skk) * 64 + scol];
      float4 w0 = wp[0], w1 = wp[1];
      unsigned short* d = &Wl[scol * 32 + skk];
      d[0 * 32] = f2bf(w0.x); d[1 * 32] = f2bf(w0.y);
      d[2 * 32] = f2bf(w0.z); d[3 * 32] = f2bf(w0.w);
      d[4 * 32] = f2bf(w1.x); d[5 * 32] = f2bf(w1.y);
      d[6 * 32] = f2bf(w1.z); d[7 * 32] = f2bf(w1.w);
    }
    __syncthreads();

    // A operand: 16 rows x 32 K of x, converted fp32->bf16
    const float* xr = x + (size_t)arow * E_DIM + k0 + kb;
    bf16x16 a = cvt16(xr, xr + 16);

    // hoist all four B operands, then issue the four WMMAs back-to-back
    bf16x16 bm[4];
#pragma unroll
    for (int n = 0; n < 4; ++n) {
      const unsigned short* wc = &Wl[(n * 16 + lq) * 32 + kb];
      bm[n] = ld16(wc, wc + 16);
    }
#pragma unroll
    for (int n = 0; n < 4; ++n)
      acc[n] = wmma_bf16(a, bm[n], acc[n]);
  }

  // bias add + bf16 store
#pragma unroll
  for (int n = 0; n < 4; ++n) {
    const int col = n * 16 + lq;
    const float bb = bias[col];
#pragma unroll
    for (int r = 0; r < 8; ++r) {
      const int orow = rowbase + hf * 8 + r;
      const unsigned short h = f2bf(acc[n][r] + bb);
      if (which == 2) {
        const int bidx = orow >> 12;            // batch (T=4096)
        const int tt   = orow & (T_SEQ - 1);    // token
        vto[((size_t)bidx * 64 + col) * T_SEQ + tt] = h;
      } else {
        unsigned short* o = (which == 0) ? qo : ko;
        o[(size_t)orow * 64 + col] = h;
      }
    }
  }
}

// ---------------------------------------------------------------------------
// Kernel 2: causal flash attention.  Block = 8 waves = 128 query rows;
// wave owns 16 query rows.  64-key tiles staged in LDS via the CDNA5 async
// global->LDS DMA path.  Second GEMM computed as O^T = V^T * P^T so the
// online-softmax rescale is a per-lane scalar.
// ---------------------------------------------------------------------------
__global__ __launch_bounds__(256) void flash_attn_kernel(
    const unsigned short* __restrict__ q,
    const unsigned short* __restrict__ k,
    const unsigned short* __restrict__ vT,
    float* __restrict__ out)
{
  const int bidx  = blockIdx.y;
  const int qbase = blockIdx.x * 128;

  const int tid  = threadIdx.x;
  const int wave = tid >> 5;
  const int lane = tid & 31;
  const int hf   = lane >> 4;
  const int lq   = lane & 15;
  const int kb   = hf ? 8 : 0;
  const int qrow0 = qbase + wave * 16;

  __shared__ __align__(16) unsigned short Kl[64 * 64];    // [key][d]   bf16
  __shared__ __align__(16) unsigned short Vl[64 * 64];    // [d][key]   bf16
  __shared__ __align__(16) unsigned short Pl[8][16 * 64]; // per-wave [q][key]
  __shared__ float Fl[8][16];                             // per-wave row stats

  const unsigned short* qB = q  + (size_t)bidx * T_SEQ * 64;
  const unsigned short* kB = k  + (size_t)bidx * T_SEQ * 64;
  const unsigned short* vB = vT + (size_t)bidx * 64 * T_SEQ;

  // Q A-operands (d=0..31 and d=32..63), resident all kernel
  const unsigned short* qr = qB + (size_t)(qrow0 + lq) * 64 + kb;
  bf16x16 aq0 = ld16(qr,      qr + 16);
  bf16x16 aq1 = ld16(qr + 32, qr + 48);

  f32x8 zero8 = {0.f, 0.f, 0.f, 0.f, 0.f, 0.f, 0.f, 0.f};
  f32x8 o[4] = {zero8, zero8, zero8, zero8};   // O^T subtiles (d-major)
  float mrow[8], lrow[8];
#pragma unroll
  for (int r = 0; r < 8; ++r) { mrow[r] = -3.0e38f; lrow[r] = 0.f; }

  // per-thread staging slots (2 x 16B per tile per buffer)
  const unsigned klds0 = (unsigned)(size_t)&Kl[0];
  const unsigned vlds0 = (unsigned)(size_t)&Vl[0];

  const int ntiles = (qbase >> 6) + 2;         // causal bound for this block
  for (int jt = 0; jt < ntiles; ++jt) {
    const int keybase = jt * 64;
    __syncthreads();
    // async DMA staging of K,V tiles (8 KB each), ASYNCcnt-tracked
#pragma unroll
    for (int e = 0; e < 2; ++e) {
      const int i  = tid + e * 256;            // 0..511 16B slots
      const int rr = i >> 3;
      const int pt = i & 7;
      async_ld_b128(klds0 + (unsigned)i * 16,
                    kB + (size_t)(keybase + rr) * 64 + pt * 8);
      async_ld_b128(vlds0 + (unsigned)i * 16,
                    vB + (size_t)rr * T_SEQ + keybase + pt * 8);
    }
    if (jt + 1 < ntiles) {  // hint next K tile (global_prefetch_b8)
      __builtin_prefetch(kB + (size_t)(keybase + 64) * 64 + tid * 16, 0, 1);
    }
    asm volatile("s_wait_asynccnt 0x0" ::: "memory");
    __syncthreads();

    if (keybase <= qrow0 + 15) {               // wave-uniform causal skip
      // ---- S = Q K^T (4 key subtiles x 2 K-steps) ----
      bf16x16 bk0[4], bk1[4];
#pragma unroll
      for (int n = 0; n < 4; ++n) {
        const unsigned short* kp = &Kl[(n * 16 + lq) * 64 + kb];
        bk0[n] = ld16(kp,      kp + 16);
        bk1[n] = ld16(kp + 32, kp + 48);
      }
      f32x8 s[4];
#pragma unroll
      for (int n = 0; n < 4; ++n) {
        f32x8 c = zero8;
        c = wmma_bf16(aq0, bk0[n], c);
        c = wmma_bf16(aq1, bk1[n], c);
        s[n] = c;
      }
      // scale 1/sqrt(64) + causal mask
#pragma unroll
      for (int n = 0; n < 4; ++n) {
        const int keyg = keybase + n * 16 + lq;
#pragma unroll
        for (int r = 0; r < 8; ++r) {
          const int rowg = qrow0 + hf * 8 + r;
          const float v = s[n][r] * 0.125f;
          s[n][r] = (keyg > rowg) ? -3.0e38f : v;
        }
      }
      // ---- online softmax (rows r within this lane-half) ----
      float fr[8];
#pragma unroll
      for (int r = 0; r < 8; ++r) {
        float mx = fmaxf(fmaxf(s[0][r], s[1][r]), fmaxf(s[2][r], s[3][r]));
        mx = fmaxf(mx, __shfl_xor(mx, 1, 32));
        mx = fmaxf(mx, __shfl_xor(mx, 2, 32));
        mx = fmaxf(mx, __shfl_xor(mx, 4, 32));
        mx = fmaxf(mx, __shfl_xor(mx, 8, 32));
        const float mnew = fmaxf(mrow[r], mx);
        fr[r] = __expf(mrow[r] - mnew);
        float sum = 0.f;
#pragma unroll
        for (int n = 0; n < 4; ++n) {
          const float p = __expf(s[n][r] - mnew);
          s[n][r] = p;
          sum += p;
        }
        sum += __shfl_xor(sum, 1, 32);
        sum += __shfl_xor(sum, 2, 32);
        sum += __shfl_xor(sum, 4, 32);
        sum += __shfl_xor(sum, 8, 32);
        lrow[r] = lrow[r] * fr[r] + sum;
        mrow[r] = mnew;
      }
      // P -> per-wave LDS as bf16, [qrow][key] (B-operand friendly)
#pragma unroll
      for (int n = 0; n < 4; ++n)
#pragma unroll
        for (int r = 0; r < 8; ++r)
          Pl[wave][(hf * 8 + r) * 64 + n * 16 + lq] = f2bf(s[n][r]);
      // per-q rescale factor via tiny LDS transpose (wave-private)
      if (lq == 0) {
#pragma unroll
        for (int r = 0; r < 8; ++r) Fl[wave][hf * 8 + r] = fr[r];
      }
      asm volatile("s_wait_dscnt 0x0" ::: "memory");
      const float fq = Fl[wave][lq];
#pragma unroll
      for (int m = 0; m < 4; ++m)
#pragma unroll
        for (int r = 0; r < 8; ++r)
          o[m][r] *= fq;
      // ---- O^T += V^T P^T ----
      const unsigned short* pp = &Pl[wave][lq * 64 + kb];
      bf16x16 bp0 = ld16(pp,      pp + 16);
      bf16x16 bp1 = ld16(pp + 32, pp + 48);
      bf16x16 av0[4], av1[4];
#pragma unroll
      for (int m = 0; m < 4; ++m) {
        const unsigned short* vp = &Vl[(m * 16 + lq) * 64 + kb];
        av0[m] = ld16(vp,      vp + 16);
        av1[m] = ld16(vp + 32, vp + 48);
      }
#pragma unroll
      for (int m = 0; m < 4; ++m) {
        o[m] = wmma_bf16(av0[m], bp0, o[m]);
        o[m] = wmma_bf16(av1[m], bp1, o[m]);
      }
    }
  }

  // ---- normalize by row sum and store fp32 output ----
  if (lq == 0) {
#pragma unroll
    for (int r = 0; r < 8; ++r) Fl[wave][hf * 8 + r] = lrow[r];
  }
  asm volatile("s_wait_dscnt 0x0" ::: "memory");
  const float inv = 1.0f / Fl[wave][lq];
  float* op = out + ((size_t)bidx * T_SEQ + qrow0 + lq) * 64;
#pragma unroll
  for (int m = 0; m < 4; ++m) {
    float4 s0 = make_float4(o[m][0] * inv, o[m][1] * inv,
                            o[m][2] * inv, o[m][3] * inv);
    float4 s1 = make_float4(o[m][4] * inv, o[m][5] * inv,
                            o[m][6] * inv, o[m][7] * inv);
    *(float4*)(op + m * 16 + hf * 8)     = s0;
    *(float4*)(op + m * 16 + hf * 8 + 4) = s1;
  }
}

// ---------------------------------------------------------------------------
extern "C" void kernel_launch(void* const* d_in, const int* in_sizes, int n_in,
                              void* d_out, int out_size, void* d_ws, size_t ws_size,
                              hipStream_t stream) {
  const float* x  = (const float*)d_in[0];
  const float* Wq = (const float*)d_in[1];
  const float* bq = (const float*)d_in[2];
  const float* Wk = (const float*)d_in[3];
  const float* bk = (const float*)d_in[4];
  const float* Wv = (const float*)d_in[5];
  const float* bv = (const float*)d_in[6];
  float* out = (float*)d_out;

  // workspace: q bf16 [B*T,64], k bf16 [B*T,64], vT bf16 [B,64,T]  (6 MB)
  unsigned short* qb  = (unsigned short*)d_ws;
  unsigned short* kbm = qb  + (size_t)B_BATCH * T_SEQ * 64;
  unsigned short* vTb = kbm + (size_t)B_BATCH * T_SEQ * 64;

  qkv_proj_kernel<<<dim3((B_BATCH * T_SEQ) / 128, 3), 256, 0, stream>>>(
      x, Wq, bq, Wk, bk, Wv, bv, qb, kbm, vTb);
  flash_attn_kernel<<<dim3(T_SEQ / 128, B_BATCH), 256, 0, stream>>>(
      qb, kbm, vTb, out);
}
